// HybridGNNv2_47459388621467
// MI455X (gfx1250) — matrix operand order repaired and checked by reading, not compiled, not run on hardware
//
#include <hip/hip_runtime.h>
#include <hip/hip_bf16.h>

typedef __attribute__((ext_vector_type(16))) _Float16 v16h;
typedef __attribute__((ext_vector_type(8)))  float    v8f;

#define BN_SCALE 0.99999500003749981f  /* 1/sqrt(1+1e-5) */

static __device__ __forceinline__ void atomicMaxF(float* addr, float val) {
    unsigned int* ua = (unsigned int*)addr;
    unsigned int old = __float_as_uint(*addr);
    while (__uint_as_float(old) < val) {
        unsigned int assumed = old;
        old = atomicCAS(ua, assumed, __float_as_uint(val));
        if (old == assumed) break;
    }
}

// ---------------------------------------------------------------------------
// Pack B (K x 128, f32 row-major) into fragment-major f16:
// chunk (tileCol, kblk) = 32 lanes x 16 halves, laid out contiguously so the
// GEMM reads each lane's B fragment as two b128 loads.
// B-fragment layout (16-bit B 32x16): lane = col n (lane&15), K = hi*8+2v+j (+8 if v>=4).
// ---------------------------------------------------------------------------
__global__ __launch_bounds__(256)
void pack_b_f16(const float* __restrict__ B, _Float16* __restrict__ out, int K)
{
    int wid  = blockIdx.x * (blockDim.x >> 5) + (threadIdx.x >> 5);
    int lane = threadIdx.x & 31;
    int nkb = K >> 5;
    if (wid >= 8 * nkb) return;
    int tileCol = wid / nkb;
    int kblk    = wid % nkb;
    int col0 = tileCol * 16;
    int k0   = kblk * 32;
    int m = lane & 15, hi = lane >> 4;
    _Float16* o = out + (size_t)wid * 512 + lane * 16;
    #pragma unroll
    for (int v = 0; v < 8; ++v)
        #pragma unroll
        for (int j = 0; j < 2; ++j) {
            int kk = hi * 8 + 2 * v + j + (v >= 4 ? 8 : 0);
            o[2 * v + j] = (_Float16)B[(size_t)(k0 + kk) * 128 + col0 + m];
        }
}

// ---------------------------------------------------------------------------
// WMMA GEMM: C[nRows x 128] = act(A[nRows x K] (+A2) @ B[K x 128] + bias (+C))
// One wave computes a 16x64 C strip = 4x v_wmma_f32_16x16x32_f16 per k-step.
// A fragment (16x32 f16): lane = row m; two contiguous 8-float runs:
//   af[0..7]  = A[row, k0 + hi*8 + 0..7]
//   af[8..15] = A[row, k0 + 16 + hi*8 + 0..7]
// ---------------------------------------------------------------------------
template <bool HAS_A2>
__global__ __launch_bounds__(256)
void wmma_gemm_f16(const float* __restrict__ A, const float* __restrict__ A2,
                   const _Float16* __restrict__ Bp, const float* __restrict__ bias,
                   float* __restrict__ C,
                   int nRows, int K, int lda, int ldc, int accum, int do_relu)
{
    int wid  = blockIdx.x * (blockDim.x >> 5) + (threadIdx.x >> 5);
    int lane = threadIdx.x & 31;
    int tileRow = wid >> 1;          // 16-row strip
    int half    = wid & 1;           // which 64-column half
    int row0 = tileRow * 16;
    if (row0 >= nRows) return;       // wave-uniform: EXEC stays all-1s
    int m = lane & 15, hi = lane >> 4;
    int nkb = K >> 5;

    const float* Arow  = A + (size_t)(row0 + m) * lda + hi * 8;
    const float* A2row = HAS_A2 ? (A2 + (size_t)(row0 + m) * lda + hi * 8) : nullptr;
    const _Float16* bbase = Bp + (size_t)(half * 4) * nkb * 512 + lane * 16;

    v8f accs[4] = {};
    for (int kb = 0; kb < nkb; ++kb) {
        const float* ap = Arow + kb * 32;
        float4 q0 = *(const float4*)(ap + 0);
        float4 q1 = *(const float4*)(ap + 4);
        float4 q2 = *(const float4*)(ap + 16);
        float4 q3 = *(const float4*)(ap + 20);
        if (HAS_A2) {
            const float* a2p = A2row + kb * 32;
            float4 p0 = *(const float4*)(a2p + 0);
            float4 p1 = *(const float4*)(a2p + 4);
            float4 p2 = *(const float4*)(a2p + 16);
            float4 p3 = *(const float4*)(a2p + 20);
            q0.x += p0.x; q0.y += p0.y; q0.z += p0.z; q0.w += p0.w;
            q1.x += p1.x; q1.y += p1.y; q1.z += p1.z; q1.w += p1.w;
            q2.x += p2.x; q2.y += p2.y; q2.z += p2.z; q2.w += p2.w;
            q3.x += p3.x; q3.y += p3.y; q3.z += p3.z; q3.w += p3.w;
        }
        v16h af;
        af[0]  = (_Float16)q0.x; af[1]  = (_Float16)q0.y;
        af[2]  = (_Float16)q0.z; af[3]  = (_Float16)q0.w;
        af[4]  = (_Float16)q1.x; af[5]  = (_Float16)q1.y;
        af[6]  = (_Float16)q1.z; af[7]  = (_Float16)q1.w;
        af[8]  = (_Float16)q2.x; af[9]  = (_Float16)q2.y;
        af[10] = (_Float16)q2.z; af[11] = (_Float16)q2.w;
        af[12] = (_Float16)q3.x; af[13] = (_Float16)q3.y;
        af[14] = (_Float16)q3.z; af[15] = (_Float16)q3.w;

        #pragma unroll
        for (int c = 0; c < 4; ++c) {
            v16h bf = *(const v16h*)(bbase + (size_t)(c * nkb + kb) * 512);
            accs[c] = __builtin_amdgcn_wmma_f32_16x16x32_f16(false, af, false, bf,
                                                             (short)0, accs[c],
                                                             false, false);
        }
    }
    // Epilogue: C/D layout VGPR r -> row hi*8+r, lane -> col n = lane&15
    #pragma unroll
    for (int c = 0; c < 4; ++c) {
        int col0 = half * 64 + c * 16;
        float bv = bias ? bias[col0 + m] : 0.0f;
        #pragma unroll
        for (int r = 0; r < 8; ++r) {
            size_t idx = (size_t)(row0 + hi * 8 + r) * ldc + col0 + m;
            float v = accs[c][r] + bv;
            if (accum)   v += C[idx];
            if (do_relu) v  = fmaxf(v, 0.0f);
            C[idx] = v;
        }
    }
}

// ---------------------------------------------------------------------------
// Utility / elementwise kernels
// ---------------------------------------------------------------------------
__global__ void fill_f32(float* p, float v, long long n) {
    long long i = (long long)blockIdx.x * blockDim.x + threadIdx.x;
    long long s = (long long)gridDim.x * blockDim.x;
    for (; i < n; i += s) p[i] = v;
}

__global__ void deg_count(const int* __restrict__ dst, int E, float* __restrict__ deg) {
    int i = blockIdx.x * blockDim.x + threadIdx.x;
    if (i < E) atomicAdd(&deg[dst[i]], 1.0f);
}

__global__ void rsqrt_inplace(float* p, int n) {
    int i = blockIdx.x * blockDim.x + threadIdx.x;
    if (i < n) p[i] = rsqrtf(p[i]);
}

// wave-per-edge: out[dst, :128] += h[src, :128] * dinv[src]*dinv[dst]
__global__ __launch_bounds__(256)
void gcn_scatter(const float* __restrict__ h, const float* __restrict__ dinv,
                 const int* __restrict__ src, const int* __restrict__ dst,
                 int E, float* __restrict__ out)
{
    int t = blockIdx.x * blockDim.x + threadIdx.x;
    int e = t >> 5, lane = t & 31;
    if (e >= E) return;
    int s = src[e], d = dst[e];
    float nrm = dinv[s] * dinv[d];
    const float4 v = *(const float4*)(h + (size_t)s * 128 + lane * 4);
    float* o = out + (size_t)d * 128 + lane * 4;
    atomicAdd(o + 0, v.x * nrm); atomicAdd(o + 1, v.y * nrm);
    atomicAdd(o + 2, v.z * nrm); atomicAdd(o + 3, v.w * nrm);
}

// stems[n, f] = relu(acc[n,f] + h[n, f]*dinv[n]^2 + b[f])
__global__ void gcn_finish(const float* __restrict__ acc, const float* __restrict__ h,
                           const float* __restrict__ dinv, const float* __restrict__ b,
                           float* __restrict__ stems, long long total)
{
    long long i = (long long)blockIdx.x * blockDim.x + threadIdx.x;
    if (i >= total) return;
    int n = (int)(i >> 7), f = (int)(i & 127);
    float di = dinv[n];
    float v = acc[i] + h[i] * di * di + b[f];
    stems[(size_t)n * 384 + f] = fmaxf(v, 0.0f);
}

// wave-per-edge: out[dst, :128] += feat[src, :128]
__global__ __launch_bounds__(256)
void scatter_add_feat(const float* __restrict__ feat, const int* __restrict__ src,
                      const int* __restrict__ dst, int E, float* __restrict__ out)
{
    int t = blockIdx.x * blockDim.x + threadIdx.x;
    int e = t >> 5, lane = t & 31;
    if (e >= E) return;
    int s = src[e], d = dst[e];
    const float4 v = *(const float4*)(feat + (size_t)s * 128 + lane * 4);
    float* o = out + (size_t)d * 128 + lane * 4;
    atomicAdd(o + 0, v.x); atomicAdd(o + 1, v.y);
    atomicAdd(o + 2, v.z); atomicAdd(o + 3, v.w);
}

// per (node, head): al_s / al_d
__global__ void gat_al(const float* __restrict__ xp, const float* __restrict__ a_src,
                       const float* __restrict__ a_dst, float* __restrict__ al_s,
                       float* __restrict__ al_d, int N)
{
    int idx = blockIdx.x * blockDim.x + threadIdx.x;
    if (idx >= N * 4) return;
    int n = idx >> 2, h = idx & 3;
    const float* p = xp + (size_t)n * 128 + h * 32;
    float as = 0.f, ad = 0.f;
    for (int k = 0; k < 32; ++k) {
        float v = p[k];
        as += v * a_src[h * 32 + k];
        ad += v * a_dst[h * 32 + k];
    }
    al_s[idx] = as; al_d[idx] = ad;
}

__device__ __forceinline__ float leaky02(float z) { return z > 0.f ? z : 0.2f * z; }

__global__ void gat_edge_max(const int* __restrict__ src, const int* __restrict__ dst,
                             const float* __restrict__ al_s, const float* __restrict__ al_d,
                             float* __restrict__ mbuf, int count, int selfloop)
{
    int idx = blockIdx.x * blockDim.x + threadIdx.x;
    if (idx >= count * 4) return;
    int e = idx >> 2, h = idx & 3;
    int s = selfloop ? e : src[e];
    int d = selfloop ? e : dst[e];
    float z = leaky02(al_s[s * 4 + h] + al_d[d * 4 + h]);
    atomicMaxF(&mbuf[d * 4 + h], z);
}

__global__ void gat_edge_sum(const int* __restrict__ src, const int* __restrict__ dst,
                             const float* __restrict__ al_s, const float* __restrict__ al_d,
                             const float* __restrict__ mbuf, float* __restrict__ sbuf,
                             int count, int selfloop)
{
    int idx = blockIdx.x * blockDim.x + threadIdx.x;
    if (idx >= count * 4) return;
    int e = idx >> 2, h = idx & 3;
    int s = selfloop ? e : src[e];
    int d = selfloop ? e : dst[e];
    float z = leaky02(al_s[s * 4 + h] + al_d[d * 4 + h]);
    atomicAdd(&sbuf[d * 4 + h], __expf(z - mbuf[d * 4 + h]));
}

// wave-per-edge weighted scatter: out[dst] += alpha(head) * xp[src]
__global__ __launch_bounds__(256)
void gat_scatter(const int* __restrict__ src, const int* __restrict__ dst,
                 const float* __restrict__ xp, const float* __restrict__ al_s,
                 const float* __restrict__ al_d, const float* __restrict__ mbuf,
                 const float* __restrict__ sbuf, float* __restrict__ out,
                 int count, int selfloop)
{
    int t = blockIdx.x * blockDim.x + threadIdx.x;
    int e = t >> 5, lane = t & 31;
    if (e >= count) return;
    int s = selfloop ? e : src[e];
    int d = selfloop ? e : dst[e];
    int h = lane >> 3;                       // 4 feats/lane, 8 lanes/head
    float z = leaky02(al_s[s * 4 + h] + al_d[d * 4 + h]);
    float alpha = __expf(z - mbuf[d * 4 + h]) / sbuf[d * 4 + h];
    const float4 v = *(const float4*)(xp + (size_t)s * 128 + lane * 4);
    float* o = out + (size_t)d * 128 + lane * 4;
    atomicAdd(o + 0, v.x * alpha); atomicAdd(o + 1, v.y * alpha);
    atomicAdd(o + 2, v.z * alpha); atomicAdd(o + 3, v.w * alpha);
}

__global__ void bias_relu_128(const float* __restrict__ in, const float* __restrict__ b,
                              float* __restrict__ out, long long total)
{
    long long i = (long long)blockIdx.x * blockDim.x + threadIdx.x;
    if (i >= total) return;
    int f = (int)(i & 127);
    out[i] = fmaxf(in[i] + b[f], 0.0f);
}

// wave-per-node: gate logits (128->3), softmax, fuse, BN0, relu -> xx
__global__ __launch_bounds__(256)
void gate_combine(const float* __restrict__ stems, const float* __restrict__ hidden,
                  const float* __restrict__ gW2, const float* __restrict__ gb2,
                  const float* __restrict__ bn0g, const float* __restrict__ bn0b,
                  float* __restrict__ xx, int N)
{
    int t = blockIdx.x * blockDim.x + threadIdx.x;
    int n = t >> 5, lane = t & 31;
    if (n >= N) return;
    const float* hr = hidden + (size_t)n * 128;
    float p0 = 0.f, p1 = 0.f, p2 = 0.f;
    #pragma unroll
    for (int k = 0; k < 4; ++k) {
        int f = lane + k * 32;
        float hv = hr[f];
        p0 += hv * gW2[f * 3 + 0];
        p1 += hv * gW2[f * 3 + 1];
        p2 += hv * gW2[f * 3 + 2];
    }
    for (int off = 16; off; off >>= 1) {
        p0 += __shfl_xor(p0, off, 32);
        p1 += __shfl_xor(p1, off, 32);
        p2 += __shfl_xor(p2, off, 32);
    }
    float l0 = p0 + gb2[0], l1 = p1 + gb2[1], l2 = p2 + gb2[2];
    float mx = fmaxf(l0, fmaxf(l1, l2));
    float e0 = __expf(l0 - mx), e1 = __expf(l1 - mx), e2 = __expf(l2 - mx);
    float inv = 1.0f / (e0 + e1 + e2);
    float g0 = e0 * inv, g1 = e1 * inv, g2 = e2 * inv;
    const float* sr = stems + (size_t)n * 384;
    float* xr = xx + (size_t)n * 128;
    #pragma unroll
    for (int k = 0; k < 4; ++k) {
        int f = lane + k * 32;
        float v = g0 * sr[f] + g1 * sr[128 + f] + g2 * sr[256 + f];
        v = v * (bn0g[f] * BN_SCALE) + bn0b[f];
        xr[f] = fmaxf(v, 0.0f);
    }
}

// xx = relu(bn(h)) + xx ; also write into reps slice (ld 384)
__global__ void conv_post(const float* __restrict__ h, const float* __restrict__ bng,
                          const float* __restrict__ bnb, float* __restrict__ xx,
                          float* __restrict__ reps_slice, long long total)
{
    long long i = (long long)blockIdx.x * blockDim.x + threadIdx.x;
    if (i >= total) return;
    int f = (int)(i & 127);
    float v = h[i] * (bng[f] * BN_SCALE) + bnb[f];
    v = fmaxf(v, 0.0f) + xx[i];
    xx[i] = v;
    reps_slice[(i >> 7) * 384 + f] = v;
}

// wave-per-node: a_log[n] = dot(hidden[n,:128], w) + b
__global__ __launch_bounds__(256)
void att_logit(const float* __restrict__ hid, const float* __restrict__ w,
               const float* __restrict__ b, float* __restrict__ alog, int N)
{
    int t = blockIdx.x * blockDim.x + threadIdx.x;
    int n = t >> 5, lane = t & 31;
    if (n >= N) return;
    const float* hr = hid + (size_t)n * 128;
    float p = 0.f;
    #pragma unroll
    for (int k = 0; k < 4; ++k) { int f = lane + k * 32; p += hr[f] * w[f]; }
    for (int off = 16; off; off >>= 1) p += __shfl_xor(p, off, 32);
    if (lane == 0) alog[n] = p + b[0];
}

__global__ void graph_max(const float* __restrict__ alog, float* __restrict__ gmax,
                          int N, int npg)
{
    int i = blockIdx.x * blockDim.x + threadIdx.x;
    if (i < N) atomicMaxF(&gmax[i / npg], alog[i]);
}

__global__ void graph_sum(const float* __restrict__ alog, const float* __restrict__ gmax,
                          float* __restrict__ gsum, int N, int npg)
{
    int i = blockIdx.x * blockDim.x + threadIdx.x;
    if (i < N) atomicAdd(&gsum[i / npg], __expf(alog[i] - gmax[i / npg]));
}

// block g, thread f: mean/max/add/att pooling over contiguous npg nodes
__global__ __launch_bounds__(128)
void pool_kernel(const float* __restrict__ xx, const float* __restrict__ alog,
                 const float* __restrict__ gmax, const float* __restrict__ gsum,
                 float* __restrict__ pooled, int npg)
{
    int g = blockIdx.x, f = threadIdx.x;
    float sum = 0.f, mx = -3.4e38f, att = 0.f;
    float gm = gmax[g], inv = 1.0f / gsum[g];
    int base = g * npg;
    for (int i = 0; i < npg; ++i) {
        int n = base + i;
        float v = xx[(size_t)n * 128 + f];
        sum += v;
        mx = fmaxf(mx, v);
        att += v * (__expf(alog[n] - gm) * inv);
    }
    pooled[g * 512 + f]       = sum / (float)npg;
    pooled[g * 512 + 128 + f] = mx;
    pooled[g * 512 + 256 + f] = sum;
    pooled[g * 512 + 384 + f] = att;
}

__global__ __launch_bounds__(128)
void head1(const float* __restrict__ pooled, const float* __restrict__ W,
           const float* __restrict__ b, float* __restrict__ hg)
{
    int g = blockIdx.x, f = threadIdx.x;
    float acc = b[f];
    for (int k = 0; k < 512; ++k) acc += pooled[g * 512 + k] * W[k * 128 + f];
    hg[g * 128 + f] = acc;
}

__global__ __launch_bounds__(64)
void head2(const float* __restrict__ hg, const float* __restrict__ W,
           const float* __restrict__ b, float* __restrict__ t1)
{
    int g = blockIdx.x, f = threadIdx.x;
    float acc = b[f];
    for (int k = 0; k < 128; ++k) acc += hg[g * 128 + k] * W[k * 64 + f];
    t1[g * 64 + f] = fmaxf(acc, 0.0f);
}

__global__ void head3(const float* __restrict__ t1, const float* __restrict__ W,
                      const float* __restrict__ b, float* __restrict__ out, int total)
{
    int idx = blockIdx.x * blockDim.x + threadIdx.x;
    if (idx >= total) return;
    int g = idx >> 1, c = idx & 1;
    float acc = b[c];
    for (int k = 0; k < 64; ++k) acc += t1[g * 64 + k] * W[k * 2 + c];
    out[idx] = acc;
}

// ---------------------------------------------------------------------------
static inline int cdiv(long long a, long long b) { return (int)((a + b - 1) / b); }

extern "C" void kernel_launch(void* const* d_in, const int* in_sizes, int n_in,
                              void* d_out, int out_size, void* d_ws, size_t ws_size,
                              hipStream_t stream)
{
    const int N = in_sizes[0] / 128;
    const int E = in_sizes[1] / 2;
    const int NG = 50;
    const int npg = N / NG;

    const float* x        = (const float*)d_in[0];
    const int*   ei       = (const int*)  d_in[1];
    const int*   src      = ei;
    const int*   dst      = ei + E;
    const float* W_gcn    = (const float*)d_in[3];
    const float* b_gcn    = (const float*)d_in[4];
    const float* gin_W1   = (const float*)d_in[5];
    const float* gin_b1   = (const float*)d_in[6];
    const float* gin_W2   = (const float*)d_in[7];
    const float* gin_b2   = (const float*)d_in[8];
    const float* W_gat    = (const float*)d_in[9];
    const float* b_gat    = (const float*)d_in[10];
    const float* a_src    = (const float*)d_in[11];
    const float* a_dst    = (const float*)d_in[12];
    const float* gproj_W  = (const float*)d_in[13];
    const float* gproj_b  = (const float*)d_in[14];
    const float* gate_W1  = (const float*)d_in[15];
    const float* gate_b1  = (const float*)d_in[16];
    const float* gate_W2  = (const float*)d_in[17];
    const float* gate_b2  = (const float*)d_in[18];
    const float* bn0_g    = (const float*)d_in[19];
    const float* bn0_b    = (const float*)d_in[20];
    const float* crel_W   = (const float*)d_in[21];
    const float* crel_b   = (const float*)d_in[22];
    const float* croot_W  = (const float*)d_in[23];
    const float* bn_g     = (const float*)d_in[24];
    const float* bn_b     = (const float*)d_in[25];
    const float* jump_W   = (const float*)d_in[26];
    const float* jump_b   = (const float*)d_in[27];
    const float* att_W1   = (const float*)d_in[28];
    const float* att_b1   = (const float*)d_in[29];
    const float* att_W2   = (const float*)d_in[30];
    const float* att_b2   = (const float*)d_in[31];
    const float* comb_W   = (const float*)d_in[32];
    const float* comb_b   = (const float*)d_in[33];
    const float* cls_W1   = (const float*)d_in[34];
    const float* cls_b1   = (const float*)d_in[35];
    const float* cls_W2   = (const float*)d_in[36];
    const float* cls_b2   = (const float*)d_in[37];
    float* out = (float*)d_out;

    // ---- workspace carve-up (256B aligned regions) ----
    char* wsb = (char*)d_ws;
    auto allocB = [&](size_t bytes) {
        void* p = (void*)wsb;
        wsb += ((bytes + 255) / 256) * 256;
        return p;
    };
    float* stems  = (float*)allocB((size_t)N * 384 * 4);  // [x_gcn | x_gin | x_gat]
    float* bufh   = (float*)allocB((size_t)N * 128 * 4);  // GEMM temp / MLP hidden
    float* aggr   = (float*)allocB((size_t)N * 128 * 4);  // shared edge accumulator
    float* xp     = (float*)allocB((size_t)N * 128 * 4);  // GAT projected features
    float* xx     = (float*)allocB((size_t)N * 128 * 4);
    float* reps   = (float*)allocB((size_t)N * 384 * 4);
    float* deg    = (float*)allocB((size_t)N * 4);        // becomes dinv in place
    float* al_s   = (float*)allocB((size_t)N * 4 * 4);
    float* al_d   = (float*)allocB((size_t)N * 4 * 4);
    float* mbuf   = (float*)allocB((size_t)N * 4 * 4);
    float* sbuf   = (float*)allocB((size_t)N * 4 * 4);
    float* alog   = (float*)allocB((size_t)N * 4);
    float* gmax   = (float*)allocB(64 * 4);
    float* gsum   = (float*)allocB(64 * 4);
    float* pooled = (float*)allocB((size_t)NG * 512 * 4);
    float* hg     = (float*)allocB((size_t)NG * 128 * 4);
    float* t1     = (float*)allocB((size_t)NG * 64 * 4);
    _Float16* packB = (_Float16*)allocB((size_t)512 * 128 * 2);  // max-K packed B

    const long long NF = (long long)N * 128;
    const int TB = 256;
    auto gemm = [&](const float* A, const float* A2, const float* B, const float* bias,
                    float* C, int K, int lda, int ldc, int accum, int relu_) {
        int nkb = K / 32;
        pack_b_f16<<<cdiv((long long)8 * nkb * 32, TB), TB, 0, stream>>>(B, packB, K);
        int waves  = (N / 16) * 2;      // 16x64 strip per wave
        int blocks = cdiv((long long)waves * 32, TB);
        if (A2)
            wmma_gemm_f16<true><<<blocks, TB, 0, stream>>>(A, A2, packB, bias, C,
                                                           N, K, lda, ldc, accum, relu_);
        else
            wmma_gemm_f16<false><<<blocks, TB, 0, stream>>>(A, nullptr, packB, bias, C,
                                                            N, K, lda, ldc, accum, relu_);
    };
    auto fill = [&](float* p, float v, long long n) {
        fill_f32<<<2048, TB, 0, stream>>>(p, v, n);
    };

    // ---- degrees (self-loop => init 1.0) ----
    fill(deg, 1.0f, N);
    deg_count<<<cdiv(E, TB), TB, 0, stream>>>(dst, E, deg);
    rsqrt_inplace<<<cdiv(N, TB), TB, 0, stream>>>(deg, N);   // deg := dinv

    // ---- GCN stem ----
    gemm(x, nullptr, W_gcn, nullptr, bufh, 128, 128, 128, 0, 0);     // h = x@W_gcn
    fill(aggr, 0.0f, NF);
    gcn_scatter<<<cdiv((long long)E * 32, TB), TB, 0, stream>>>(bufh, deg, src, dst, E, aggr);
    gcn_finish<<<cdiv(NF, TB), TB, 0, stream>>>(aggr, bufh, deg, b_gcn, stems, NF);

    // ---- GIN stem ----
    fill(aggr, 0.0f, NF);
    scatter_add_feat<<<cdiv((long long)E * 32, TB), TB, 0, stream>>>(x, src, dst, E, aggr);
    gemm(x, aggr, gin_W1, gin_b1, bufh, 128, 128, 128, 0, 1);        // relu((x+aggr)W1+b1)
    gemm(bufh, nullptr, gin_W2, gin_b2, stems + 128, 128, 128, 384, 0, 1);

    // ---- GAT stem ----
    gemm(x, nullptr, W_gat, nullptr, xp, 128, 128, 128, 0, 0);       // xp = x@W_gat
    gat_al<<<cdiv((long long)N * 4, TB), TB, 0, stream>>>(xp, a_src, a_dst, al_s, al_d, N);
    fill(mbuf, -3.4e38f, (long long)N * 4);
    gat_edge_max<<<cdiv((long long)E * 4, TB), TB, 0, stream>>>(src, dst, al_s, al_d, mbuf, E, 0);
    gat_edge_max<<<cdiv((long long)N * 4, TB), TB, 0, stream>>>(src, dst, al_s, al_d, mbuf, N, 1);
    fill(sbuf, 0.0f, (long long)N * 4);
    gat_edge_sum<<<cdiv((long long)E * 4, TB), TB, 0, stream>>>(src, dst, al_s, al_d, mbuf, sbuf, E, 0);
    gat_edge_sum<<<cdiv((long long)N * 4, TB), TB, 0, stream>>>(src, dst, al_s, al_d, mbuf, sbuf, N, 1);
    fill(aggr, 0.0f, NF);
    gat_scatter<<<cdiv((long long)E * 32, TB), TB, 0, stream>>>(src, dst, xp, al_s, al_d,
                                                                mbuf, sbuf, aggr, E, 0);
    gat_scatter<<<cdiv((long long)N * 32, TB), TB, 0, stream>>>(src, dst, xp, al_s, al_d,
                                                                mbuf, sbuf, aggr, N, 1);
    bias_relu_128<<<cdiv(NF, TB), TB, 0, stream>>>(aggr, b_gat, xp, NF);  // relu(x_gat+b)
    gemm(xp, nullptr, gproj_W, gproj_b, stems + 256, 128, 128, 384, 0, 0);

    // ---- gated fusion + BN0 + relu ----
    gemm(stems, nullptr, gate_W1, gate_b1, bufh, 384, 384, 128, 0, 1);
    gate_combine<<<cdiv((long long)N * 32, TB), TB, 0, stream>>>(stems, bufh, gate_W2,
                                                                 gate_b2, bn0_g, bn0_b, xx, N);

    // ---- GraphConv stack (L=3) with residuals, write reps ----
    for (int i = 0; i < 3; ++i) {
        fill(aggr, 0.0f, NF);
        scatter_add_feat<<<cdiv((long long)E * 32, TB), TB, 0, stream>>>(xx, src, dst, E, aggr);
        gemm(aggr, nullptr, crel_W + (size_t)i * 128 * 128, crel_b + i * 128,
             bufh, 128, 128, 128, 0, 0);
        gemm(xx, nullptr, croot_W + (size_t)i * 128 * 128, nullptr,
             bufh, 128, 128, 128, 1, 0);                           // accumulate
        conv_post<<<cdiv(NF, TB), TB, 0, stream>>>(bufh, bn_g + i * 128, bn_b + i * 128,
                                                   xx, reps + i * 128, NF);
    }

    // ---- jumping knowledge (cat) projection ----
    gemm(reps, nullptr, jump_W, jump_b, xx, 384, 384, 128, 0, 0);

    // ---- attention logits ----
    gemm(xx, nullptr, att_W1, att_b1, bufh, 128, 128, 128, 0, 1);
    att_logit<<<cdiv((long long)N * 32, TB), TB, 0, stream>>>(bufh, att_W2, att_b2, alog, N);

    // ---- global pooling (mean/max/add/att) ----
    fill(gmax, -3.4e38f, 64);
    fill(gsum, 0.0f, 64);
    graph_max<<<cdiv(N, TB), TB, 0, stream>>>(alog, gmax, N, npg);
    graph_sum<<<cdiv(N, TB), TB, 0, stream>>>(alog, gmax, gsum, N, npg);
    pool_kernel<<<NG, 128, 0, stream>>>(xx, alog, gmax, gsum, pooled, npg);

    // ---- classification head ----
    head1<<<NG, 128, 0, stream>>>(pooled, comb_W, comb_b, hg);
    head2<<<NG, 64, 0, stream>>>(hg, cls_W1, cls_b1, t1);
    head3<<<1, 128, 0, stream>>>(t1, cls_W2, cls_b2, out, NG * 2);

    (void)n_in; (void)out_size; (void)ws_size;
}
